// LengthRegulator_81235011436663
// MI455X (gfx1250) — compile-verified, matrix operand order checked by reading, not run on hardware
//
#include <hip/hip_runtime.h>

// ---------------------------------------------------------------------------
// FastSpeech duration predictor + length regulator for MI455X (gfx1250).
//  - conv1d(k=3) = 3 shifted GEMMs on v_wmma_f32_16x16x32_bf16
//  - weights streamed via global_load_async_to_lds_b128 into a 3-deep LDS
//    ring (ASYNCcnt-tracked, prefetch distance 2)
//  - A fragments software-pipelined from a zero-padded bf16 activation buffer
//  - LayerNorm+ReLU fused via wave32 __shfl_xor reductions, final F->1 linear
//    fused into conv2 epilogue; LR is a float4 gather kernel.
// ---------------------------------------------------------------------------

typedef __attribute__((ext_vector_type(16))) __bf16       v16bf;
typedef __attribute__((ext_vector_type(8)))  float        v8f;
typedef __attribute__((ext_vector_type(4)))  unsigned int u32x4;

union Frag16 { u32x4 q[2]; v16bf v; };

#define TT      512
#define PT      514              // padded time (1 zero row each side)
#define CC      256
#define BBATCH  32
#define MEL     2304
#define OUT_ELEMS (BBATCH * MEL * CC)   // 18,874,368
#define KSTEPS  24               // 3 shifts x 8 k-chunks of 32

__device__ __forceinline__ unsigned short f2bf(float f) {
    unsigned u = __builtin_bit_cast(unsigned, f);
    u += 0x7FFFu + ((u >> 16) & 1u);          // round-to-nearest-even
    return (unsigned short)(u >> 16);
}

// ---------------------------------------------------------------------------
// Repack conv weights [F=256][D=256][K=3] f32 -> bf16 in WMMA B-fragment
// order: slab for k-step ks=(s*8+k32) is 16KB contiguous; within it, tile nt,
// lane l owns 16 contiguous bf16 (32 B).
// ---------------------------------------------------------------------------
__global__ __launch_bounds__(256)
void repack_w_kernel(const float* __restrict__ w, unsigned short* __restrict__ out) {
    int e = blockIdx.x * 256 + threadIdx.x;
    if (e >= 256 * 256 * 3) return;
    int f = e / 768;
    int r = e % 768;
    int d = r / 3;
    int s = r % 3;
    int nt   = f >> 4, nl = f & 15;
    int k32  = d >> 5, dk = d & 31;
    int hi   = (dk >> 3) & 1;                       // lane half holding this K
    int pos  = (dk & 7) + ((dk >> 4) << 3);         // element within lane
    int lane = nl + (hi << 4);
    out[(((s * 8 + k32) * 16 + nt) * 32 + lane) * 16 + pos] = f2bf(w[e]);
}

// ---------------------------------------------------------------------------
// x f32 [B][512][256] -> bf16 padded [B][514][256], halo rows zeroed.
// ---------------------------------------------------------------------------
__global__ __launch_bounds__(256)
void convert_pad_kernel(const float* __restrict__ x, unsigned short* __restrict__ xb) {
    int idx = blockIdx.x * 256 + threadIdx.x;
    if (idx >= BBATCH * PT * CC) return;
    int bb  = idx / (PT * CC);
    int rem = idx - bb * PT * CC;
    int t   = rem / CC;
    int d   = rem - t * CC;
    unsigned short v = 0;
    if (t >= 1 && t <= TT)
        v = f2bf(x[((size_t)(bb * TT + t - 1)) * CC + d]);
    xb[idx] = v;
}

// zero the 2 halo rows per batch of a padded bf16 activation buffer
__global__ __launch_bounds__(256)
void zero_pad_kernel(unsigned short* __restrict__ h) {
    int idx = blockIdx.x * 256 + threadIdx.x;       // 32*2*256 = 16384
    int bb   = idx >> 9;
    int rem  = idx & 511;
    int prow = (rem >> 8) ? (PT - 1) : 0;
    int col  = rem & 255;
    h[((size_t)(bb * PT + prow)) * CC + col] = 0;
}

// ---------------------------------------------------------------------------
// Fused conv1d(k=3,pad=1) + bias + LayerNorm + ReLU (+ optional F->1 linear).
// Block: 256 threads = 8 waves (4 M-tiles x 2 N-halves); 64 rows x 256 chans.
// B slabs: async global->LDS, 3-deep ring, prefetch distance 2.
// ---------------------------------------------------------------------------
template<int WRITE_H, int WRITE_DUR>
__global__ __launch_bounds__(256)
void conv_ln_kernel(const unsigned short* __restrict__ in,   // bf16 padded [B][514][256]
                    const unsigned short* __restrict__ Wr,   // repacked bf16 weights
                    const float* __restrict__ bias,
                    const float* __restrict__ gln,
                    const float* __restrict__ bln,
                    const float* __restrict__ lin_w,
                    const float* __restrict__ lin_b,
                    unsigned short* __restrict__ Hout,       // bf16 padded [B][514][256]
                    float* __restrict__ durOut) {
    const int tid  = threadIdx.x;
    const int lane = tid & 31;
    const int wave = tid >> 5;
    const int wm   = wave & 3;        // M tile 0..3  (rows wm*16..+15)
    const int wn   = wave >> 2;       // N half 0..1  (cols wn*128..+127)
    const int nl   = lane & 15;
    const int hi   = lane >> 4;
    const int t0   = blockIdx.x * 64;
    const int bb   = blockIdx.y;

    __shared__ unsigned short Bslab[3][8192];                // 48 KB ring
    __shared__ float gS[256], bS[256], biasS[256], lwS[256];
    __shared__ float partS[64][2], partQ[64][2], partD[64][2];
    __shared__ float muS[64], rsS[64];

    gS[tid]    = gln[tid];
    bS[tid]    = bln[tid];
    biasS[tid] = bias[tid];
    if (WRITE_DUR) lwS[tid] = lin_w[tid];

    // cooperative async copy of one 16KB weight slab: 64 B per thread
    auto issue_slab = [&](int ks, int bufi) {
        unsigned lds = (unsigned)(size_t)(&Bslab[bufi][tid * 32]);
        const unsigned short* g = Wr + (size_t)ks * 8192 + tid * 32;
        asm volatile(
            "global_load_async_to_lds_b128 %0, %1, off\n\t"
            "global_load_async_to_lds_b128 %0, %1, off offset:16\n\t"
            "global_load_async_to_lds_b128 %0, %1, off offset:32\n\t"
            "global_load_async_to_lds_b128 %0, %1, off offset:48"
            :: "v"(lds), "v"(g) : "memory");
    };
    auto load_a = [&](int ks) -> Frag16 {
        int s = ks >> 3, k32 = ks & 7;
        int arow = t0 + wm * 16 + nl + s;        // padded-row coords
        const unsigned short* ap =
            in + ((size_t)(bb * PT + arow)) * CC + k32 * 32 + hi * 8;
        Frag16 a;
        a.q[0] = *reinterpret_cast<const u32x4*>(ap);
        a.q[1] = *reinterpret_cast<const u32x4*>(ap + 16);
        return a;
    };

    issue_slab(0, 0);
    issue_slab(1, 1);
    Frag16 acur = load_a(0);

    v8f z8 = {0.f, 0.f, 0.f, 0.f, 0.f, 0.f, 0.f, 0.f};
    v8f acc[8];
#pragma unroll
    for (int j = 0; j < 8; ++j) acc[j] = z8;

    int buf = 0;
    for (int ks = 0; ks < KSTEPS; ++ks) {
        if (ks < KSTEPS - 1)
            asm volatile("s_wait_asynccnt 0x4" ::: "memory");  // slab ks landed
        else
            asm volatile("s_wait_asynccnt 0x0" ::: "memory");
        __syncthreads();
        if (ks + 2 < KSTEPS) {
            int nb = buf + 2; if (nb >= 3) nb -= 3;
            issue_slab(ks + 2, nb);
        }
        Frag16 anext = acur;
        if (ks + 1 < KSTEPS) anext = load_a(ks + 1);           // pipelined A

        const unsigned short* sb = &Bslab[buf][(wn * 256 + lane) * 16];
#pragma unroll
        for (int j = 0; j < 8; ++j) {
            const unsigned short* wp = sb + j * 512;
            Frag16 bfrag;
            bfrag.q[0] = *reinterpret_cast<const u32x4*>(wp);
            bfrag.q[1] = *reinterpret_cast<const u32x4*>(wp + 8);
            acc[j] = __builtin_amdgcn_wmma_f32_16x16x32_bf16(
                false, acur.v, false, bfrag.v, (short)0, acc[j], false, false);
        }
        acur = anext;
        buf = (buf + 1 == 3) ? 0 : buf + 1;
    }

    // ---- LayerNorm stats: per-row sum/sumsq, reduced in 16-lane halves ----
    float ss[8], qq[8];
#pragma unroll
    for (int r = 0; r < 8; ++r) { ss[r] = 0.f; qq[r] = 0.f; }
#pragma unroll
    for (int j = 0; j < 8; ++j) {
        int n = wn * 128 + j * 16 + nl;
        float bi = biasS[n];
#pragma unroll
        for (int r = 0; r < 8; ++r) {
            float v = acc[j][r] + bi;
            ss[r] += v;
            qq[r] += v * v;
        }
    }
#pragma unroll
    for (int r = 0; r < 8; ++r) {
#pragma unroll
        for (int m = 1; m < 16; m <<= 1) {
            ss[r] += __shfl_xor(ss[r], m, 32);
            qq[r] += __shfl_xor(qq[r], m, 32);
        }
        if (nl == 0) {
            int row = wm * 16 + hi * 8 + r;
            partS[row][wn] = ss[r];
            partQ[row][wn] = qq[r];
        }
    }
    __syncthreads();
    if (tid < 64) {
        float S   = partS[tid][0] + partS[tid][1];
        float Q   = partQ[tid][0] + partQ[tid][1];
        float mu  = S * (1.f / 256.f);
        float var = Q * (1.f / 256.f) - mu * mu;
        muS[tid] = mu;
        rsS[tid] = rsqrtf(var + 1e-5f);
    }
    __syncthreads();

    // ---- apply LN + ReLU in registers; emit h (bf16) and/or dur dot ----
    float dd[8];
#pragma unroll
    for (int r = 0; r < 8; ++r) dd[r] = 0.f;
#pragma unroll
    for (int j = 0; j < 8; ++j) {
        int n = wn * 128 + j * 16 + nl;
        float bi = biasS[n], gg = gS[n], be = bS[n];
        float lw = WRITE_DUR ? lwS[n] : 0.f;
#pragma unroll
        for (int r = 0; r < 8; ++r) {
            int row = wm * 16 + hi * 8 + r;
            float v = acc[j][r] + bi;
            float h = fmaf((v - muS[row]) * rsS[row], gg, be);
            h = fmaxf(h, 0.f);
            if (WRITE_H)
                Hout[((size_t)(bb * PT + t0 + row + 1)) * CC + n] = f2bf(h);
            if (WRITE_DUR)
                dd[r] = fmaf(h, lw, dd[r]);
        }
    }
    if (WRITE_DUR) {
#pragma unroll
        for (int r = 0; r < 8; ++r) {
#pragma unroll
            for (int m = 1; m < 16; m <<= 1)
                dd[r] += __shfl_xor(dd[r], m, 32);
            if (nl == 0) {
                int row = wm * 16 + hi * 8 + r;
                partD[row][wn] = dd[r];
            }
        }
        __syncthreads();
        if (tid < 64) {
            float d = partD[tid][0] + partD[tid][1] + lin_b[0];
            durOut[bb * TT + t0 + tid] = fmaxf(d, 0.f);
        }
    }
}

// ---------------------------------------------------------------------------
// Length regulator: scan target in LDS, binary search per output row,
// float4 row gather. Grid: (MEL/256, B), 256 threads (4 rows x 64 lanes).
// ---------------------------------------------------------------------------
__global__ __launch_bounds__(256)
void lr_gather_kernel(const float* __restrict__ x,
                      const int* __restrict__ target,
                      float* __restrict__ out) {
    __shared__ int cum[TT];
    __shared__ int psum[256];
    const int tid = threadIdx.x;
    const int bb  = blockIdx.y;

    int v0 = target[bb * TT + 2 * tid];
    int v1 = target[bb * TT + 2 * tid + 1];
    psum[tid] = v0 + v1;
    __syncthreads();
#pragma unroll
    for (int off = 1; off < 256; off <<= 1) {
        int t = (tid >= off) ? psum[tid - off] : 0;
        __syncthreads();
        psum[tid] += t;
        __syncthreads();
    }
    cum[2 * tid + 1] = psum[tid];
    cum[2 * tid]     = psum[tid] - v1;
    __syncthreads();
    const int total = cum[TT - 1];

    const int grp = tid >> 6, l = tid & 63;
    for (int it = 0; it < 64; ++it) {
        int row   = it * 4 + grp;
        int t_out = blockIdx.x * 256 + row;
        int lo = 0, hi = TT;                      // searchsorted right
        while (lo < hi) {
            int mid = (lo + hi) >> 1;
            if (cum[mid] <= t_out) lo = mid + 1; else hi = mid;
        }
        int idx = lo < (TT - 1) ? lo : (TT - 1);
        float scale = (t_out < total) ? 1.f : 0.f;
        float4 v = *reinterpret_cast<const float4*>(
            x + ((size_t)(bb * TT + idx)) * CC + l * 4);
        v.x *= scale; v.y *= scale; v.z *= scale; v.w *= scale;
        *reinterpret_cast<float4*>(
            out + ((size_t)bb * MEL + t_out) * CC + l * 4) = v;
    }
}

// ---------------------------------------------------------------------------
extern "C" void kernel_launch(void* const* d_in, const int* in_sizes, int n_in,
                              void* d_out, int out_size, void* d_ws, size_t ws_size,
                              hipStream_t stream) {
    const float* x      = (const float*)d_in[0];
    const int*   target = (const int*)d_in[1];
    const float* c1w    = (const float*)d_in[3];
    const float* c1b    = (const float*)d_in[4];
    const float* ln1g   = (const float*)d_in[5];
    const float* ln1b   = (const float*)d_in[6];
    const float* c2w    = (const float*)d_in[7];
    const float* c2b    = (const float*)d_in[8];
    const float* ln2g   = (const float*)d_in[9];
    const float* ln2b   = (const float*)d_in[10];
    const float* linw   = (const float*)d_in[11];
    const float* linb   = (const float*)d_in[12];

    float* out = (float*)d_out;
    float* dur = out + OUT_ELEMS;

    // ws: W1r[384K] | W2r[384K] | Xb bf16 padded[8.03MB] | H1 bf16 padded[8.03MB]
    unsigned short* W1r = (unsigned short*)d_ws;
    unsigned short* W2r = W1r + 256 * 256 * 3;
    unsigned short* Xb  = (unsigned short*)((char*)d_ws + 786432);
    unsigned short* H1  = Xb + (size_t)BBATCH * PT * CC;

    repack_w_kernel<<<768, 256, 0, stream>>>(c1w, W1r);
    repack_w_kernel<<<768, 256, 0, stream>>>(c2w, W2r);

    int npad = BBATCH * PT * CC;
    convert_pad_kernel<<<(npad + 255) / 256, 256, 0, stream>>>(x, Xb);
    zero_pad_kernel<<<64, 256, 0, stream>>>(H1);

    conv_ln_kernel<1, 0><<<dim3(TT / 64, BBATCH), 256, 0, stream>>>(
        Xb, W1r, c1b, ln1g, ln1b, nullptr, nullptr, H1, nullptr);

    conv_ln_kernel<0, 1><<<dim3(TT / 64, BBATCH), 256, 0, stream>>>(
        H1, W2r, c2b, ln2g, ln2b, linw, linb, nullptr, dur);

    lr_gather_kernel<<<dim3(MEL / 256, BBATCH), 256, 0, stream>>>(x, target, out);
}
